// EdgeSpecificComplexModel_34016140984985
// MI455X (gfx1250) — matrix operand order repaired and checked by reading, not compiled, not run on hardware
//
#include <hip/hip_runtime.h>
#include <hip/hip_bf16.h>
#include <math.h>

typedef __attribute__((ext_vector_type(16))) __bf16 v16bf;
typedef __attribute__((ext_vector_type(8)))  float  v8f;

#define B_ 262144
#define LN_EPS 1e-5f
#define PHI_SCALE 0.13089969389957471827f /* pi/24 */

// d_ws layout (bytes):
//   [0,500)        : 125 floats of edge constants: damp[25], wdp[25], wk0[25], wk1[25], s[25]
//   [512, +32768)  : W1 packed bf16 B-fragments: 32 coltiles x 32 lanes x 16 bf16
//   [33280,+16384) : W2 packed bf16 B-fragments: 16 ktiles  x 32 lanes x 16 bf16
#define WS_W1PK 512
#define WS_W2PK (512 + 32768)

// ---------------------------------------------------------------------------
// Setup: swizzle weights into WMMA fragment layout + fold edge constants.
// B-matrix (32x16, 16-bit) fragment layout: lane holds column n = lane&15;
// element e holds K = (e/8)*16 + ((lane>=16)?8:0) + e%8.
// ---------------------------------------------------------------------------
__global__ void setup_kernel(const float* __restrict__ S,
                             const float* __restrict__ gam,
                             const float* __restrict__ dph,
                             const float* __restrict__ pw,
                             const float* __restrict__ W1,
                             const float* __restrict__ W2,
                             void* __restrict__ ws) {
  const int blk = blockIdx.x, lane = threadIdx.x;
  __bf16* w1pk = (__bf16*)((char*)ws + WS_W1PK);
  __bf16* w2pk = (__bf16*)((char*)ws + WS_W2PK);
  float*  edge = (float*)ws;
  const int kh = (lane >> 4) & 1;
  const int n  = lane & 15;
  if (blk < 32) {                      // W1: (25,512), pad K 25->32
    const int ct = blk;
    #pragma unroll
    for (int e = 0; e < 16; ++e) {
      const int k = (e < 8) ? kh * 8 + e : 16 + kh * 8 + (e - 8);
      const float v = (k < 25) ? W1[k * 512 + ct * 16 + n] : 0.f;
      w1pk[((ct * 32 + lane) << 4) + e] = (__bf16)v;
    }
  } else if (blk < 48) {               // W2: (512,4), pad N 4->16
    const int kt = blk - 32;
    #pragma unroll
    for (int e = 0; e < 16; ++e) {
      const int k = kt * 32 + ((e < 8) ? kh * 8 + e : 16 + kh * 8 + (e - 8));
      const float v = (n < 4) ? W2[k * 4 + n] : 0.f;
      w2pk[((kt * 32 + lane) << 4) + e] = (__bf16)v;
    }
  } else {                             // edge constants (i,j) flattened, K=2
    const int t = lane;
    if (t < 25) {
      const float p0 = pw[t * 2 + 0], p1 = pw[t * 2 + 1];
      const float mx = fmaxf(p0, p1);
      const float e0 = __expf(p0 - mx), e1 = __expf(p1 - mx);
      const float inv = 1.f / (e0 + e1);
      const float w0 = e0 * inv, w1v = e1 * inv;
      const float sg0 = 1.f / (1.f + __expf(-gam[t * 2 + 0]));
      const float sg1 = 1.f / (1.f + __expf(-gam[t * 2 + 1]));
      edge[t]       = w0 * sg0 + w1v * sg1;                       // damp_ij
      edge[25 + t]  = w0 * dph[t * 2 + 0] + w1v * dph[t * 2 + 1]; // w.delta_phi
      edge[50 + t]  = w0;                                         // wk0
      edge[75 + t]  = w1v;                                        // wk1
      edge[100 + t] = S[t];                                       // S_ij
    }
  }
}

// ---------------------------------------------------------------------------
// Main fused kernel: 2 waves/block, one 16-row batch tile per wave.
// ---------------------------------------------------------------------------
#define HSTR 516  /* 512 + 4 pad: kills LDS bank conflicts between row halves */

__launch_bounds__(64)
__global__ void fused_kernel(const float* __restrict__ x2,
                             const float* __restrict__ hist,
                             const float* __restrict__ phig,
                             const float* __restrict__ b1,
                             const float* __restrict__ lng,
                             const float* __restrict__ lnb,
                             const float* __restrict__ b2,
                             const void* __restrict__ ws,
                             float* __restrict__ out) {
  __shared__ __align__(16) float s_h[2][16 * HSTR];  // h staging, per wave
  __shared__ float s_ln[3 * 512];                    // ln_g | ln_b | b1
  __shared__ float s_part[2][64];                    // per-lane sum | sumsq
  __shared__ float s_stat[2][32];                    // mu[16] | rstd[16]
  __shared__ float s_par[2][64];                     // params[16][4]

  const int tid  = threadIdx.x;
  const int w    = tid >> 5;
  const int lane = tid & 31;
  const int row_base = blockIdx.x * 32 + w * 16;

  for (int i = tid; i < 512; i += 64) {
    s_ln[i]        = lng[i];
    s_ln[512 + i]  = lnb[i];
    s_ln[1024 + i] = b1[i];
  }
  __syncthreads();

  const __bf16* w1pk = (const __bf16*)((const char*)ws + WS_W1PK);
  const __bf16* w2pk = (const __bf16*)((const char*)ws + WS_W2PK);
  const float*  edge = (const float*)ws;

  // ---- GEMM1 A fragment: 16x32 bf16 from history (K 25 -> 32 zero-padded).
  // A layout: lane holds row m = lane&15; element e -> K = (e/8)*16 + 8*(lane>=16) + e%8
  const int m       = lane & 15;
  const int hiHalf  = lane >> 4;
  const int rowoff  = hiHalf * 8;
  const float* hrow = hist + (size_t)(row_base + m) * 25;
  float a[16];
  #pragma unroll
  for (int e = 0; e < 8; ++e) a[e] = hrow[hiHalf * 8 + e];      // K 0..7 / 8..15
  if (!hiHalf) {
    #pragma unroll
    for (int e = 0; e < 8; ++e) a[8 + e] = hrow[16 + e];        // K 16..23
  } else {
    a[8] = hrow[24];                                            // K 24; 25..31 pad
    #pragma unroll
    for (int e = 1; e < 8; ++e) a[8 + e] = 0.f;
  }
  v16bf afrag;
  #pragma unroll
  for (int e = 0; e < 16; ++e) afrag[e] = (__bf16)a[e];

  // ---- GEMM1: h(16x512) = A x W1, one 16x16x32 WMMA per column tile.
  for (int ct = 0; ct < 32; ++ct) {
    const v16bf bfrag = *(const v16bf*)(w1pk + ((ct * 32 + lane) << 4));
    if (ct + 1 < 32)
      __builtin_prefetch((const void*)(w1pk + (((ct + 1) * 32 + lane) << 4)), 0, 3);
    v8f c = {};
    c = __builtin_amdgcn_wmma_f32_16x16x32_bf16(false, afrag, false, bfrag,
                                                (short)0, c, false, false);
    // C/D layout: VGPR r -> row rowoff+r, column = ct*16 + (lane&15)
    const int col = ct * 16 + m;
    const float b1c = s_ln[1024 + col];
    #pragma unroll
    for (int r = 0; r < 8; ++r)
      s_h[w][(rowoff + r) * HSTR + col] = c[r] + b1c;
  }
  __syncthreads();

  // ---- LayerNorm stats: 2 lanes per row, 256 cols each.
  {
    const int srow = lane >> 1, half = lane & 1;
    const float* hp = &s_h[w][srow * HSTR + half * 256];
    float s = 0.f, s2 = 0.f;
    #pragma unroll 4
    for (int t = 0; t < 256; t += 4) {
      const float4 v = *(const float4*)(hp + t);
      s  += v.x + v.y + v.z + v.w;
      s2 += v.x * v.x + v.y * v.y + v.z * v.z + v.w * v.w;
    }
    s_part[w][lane]      = s;
    s_part[w][32 + lane] = s2;
  }
  __syncthreads();
  if (lane < 16) {
    const float sum = s_part[w][2 * lane] + s_part[w][2 * lane + 1];
    const float ssq = s_part[w][32 + 2 * lane] + s_part[w][32 + 2 * lane + 1];
    const float mu  = sum * (1.f / 512.f);
    const float var = ssq * (1.f / 512.f) - mu * mu;
    s_stat[w][lane]      = mu;
    s_stat[w][16 + lane] = rsqrtf(var + LN_EPS);
  }
  __syncthreads();

  // ---- GEMM2: params(16x16, cols 0..3 live) = LN(h) x W2, K = 512.
  const float mu   = s_stat[w][m];
  const float rstd = s_stat[w][16 + m];
  v8f c2 = {};
  for (int kt = 0; kt < 16; ++kt) {
    v16bf a2;
    const int kb = kt * 32 + hiHalf * 8;
    #pragma unroll
    for (int e = 0; e < 8; ++e) {
      const int k = kb + e;
      float v = s_h[w][m * HSTR + k];
      v = (v - mu) * rstd * s_ln[k] + s_ln[512 + k];
      a2[e] = (__bf16)fmaxf(v, 0.f);
    }
    #pragma unroll
    for (int e = 0; e < 8; ++e) {
      const int k = kb + 16 + e;
      float v = s_h[w][m * HSTR + k];
      v = (v - mu) * rstd * s_ln[k] + s_ln[512 + k];
      a2[8 + e] = (__bf16)fmaxf(v, 0.f);
    }
    const v16bf b2f = *(const v16bf*)(w2pk + ((kt * 32 + lane) << 4));
    c2 = __builtin_amdgcn_wmma_f32_16x16x32_bf16(false, a2, false, b2f,
                                                 (short)0, c2, false, false);
  }
  if (m < 4) {
    const float bb = b2[m];
    #pragma unroll
    for (int r = 0; r < 8; ++r)
      s_par[w][(rowoff + r) * 4 + m] = c2[r] + bb;
  }
  __syncthreads();

  // ---- Per-row tail: one lane per batch row.
  if (lane < 16) {
    const int b = row_base + lane;
    const float p0 = s_par[w][lane * 4 + 0];
    const float p1 = s_par[w][lane * 4 + 1];
    const float p2 = s_par[w][lane * 4 + 2];
    const float p3 = s_par[w][lane * 4 + 3];
    const float phi0 = phig[b * 2 + 0] + tanhf(p0) * PHI_SCALE;
    const float phi1 = phig[b * 2 + 1] + tanhf(p2) * PHI_SCALE;
    const float r0 = 1.f / (1.f + __expf(-p1));
    const float r1 = 1.f / (1.f + __expf(-p3));
    out[(size_t)B_ * 10 + b * 2 + 0] = phi0;
    out[(size_t)B_ * 10 + b * 2 + 1] = phi1;

    float xr[5], xi[5];
    #pragma unroll
    for (int j = 0; j < 5; ++j) {
      xr[j] = x2[b * 10 + j * 2 + 0];
      xi[j] = x2[b * 10 + j * 2 + 1];
    }
    float* agout = out + (size_t)B_ * 12 + (size_t)b * 25;
    float nr[5], ni[5];
    #pragma unroll
    for (int i = 0; i < 5; ++i) { nr[i] = 0.f; ni[i] = 0.f; }
    #pragma unroll
    for (int i = 0; i < 5; ++i) {
      #pragma unroll
      for (int j = 0; j < 5; ++j) {
        const int idx = i * 5 + j;
        const float wk0 = edge[50 + idx], wk1 = edge[75 + idx];
        const float theta = wk0 * phi0 + wk1 * phi1 + edge[25 + idx];
        const float rr    = wk0 * r0 + wk1 * r1;
        const float araw  = rr * edge[100 + idx];
        const float ag    = edge[idx] * araw / (1.f + fabsf(araw));
        float st, ct;
        __sincosf(theta, &st, &ct);
        nr[i] += ag * (ct * xr[j] - st * xi[j]);
        ni[i] += ag * (st * xr[j] + ct * xi[j]);
        agout[idx] = ag;
      }
    }
    #pragma unroll
    for (int i = 0; i < 5; ++i) {
      out[(size_t)b * 10 + i * 2 + 0] = nr[i];
      out[(size_t)b * 10 + i * 2 + 1] = ni[i];
    }
  }
}

extern "C" void kernel_launch(void* const* d_in, const int* in_sizes, int n_in,
                              void* d_out, int out_size, void* d_ws, size_t ws_size,
                              hipStream_t stream) {
  (void)in_sizes; (void)n_in; (void)out_size; (void)ws_size;
  const float* x2   = (const float*)d_in[0];   // x_2ch    (B,5,2)
  const float* hist = (const float*)d_in[1];   // history  (B,25)
  const float* phig = (const float*)d_in[2];   // phi_global (B,2)
  const float* S    = (const float*)d_in[3];   // S        (5,5)
  const float* gam  = (const float*)d_in[4];   // gamma    (5,5,2)
  const float* dph  = (const float*)d_in[5];   // delta_phi(5,5,2)
  const float* pw   = (const float*)d_in[6];   // path_weights (5,5,2)
  const float* W1   = (const float*)d_in[7];   // (25,512)
  const float* b1   = (const float*)d_in[8];   // (512,)
  const float* lng  = (const float*)d_in[9];   // (512,)
  const float* lnb  = (const float*)d_in[10];  // (512,)
  const float* W2   = (const float*)d_in[11];  // (512,4)
  const float* b2   = (const float*)d_in[12];  // (4,)

  hipLaunchKernelGGL(setup_kernel, dim3(49), dim3(32), 0, stream,
                     S, gam, dph, pw, W1, W2, d_ws);
  hipLaunchKernelGGL(fused_kernel, dim3(B_ / 32), dim3(64), 0, stream,
                     x2, hist, phig, b1, lng, lnb, b2, d_ws, (float*)d_out);
}
